// MatchingNet_65042984731022
// MI455X (gfx1250) — compile-verified, matrix operand order checked by reading, not run on hardware
//
#include <hip/hip_runtime.h>
#include <hip/hip_bf16.h>
#include <cmath>

typedef __attribute__((ext_vector_type(16))) _Float16 v16h;
typedef __attribute__((ext_vector_type(8)))  _Float16 v8h;
typedef __attribute__((ext_vector_type(8)))  float    v8f;

#define LDIM 4800
#define DDIM 256
#define KTOP 512
#define NPTS 1024   // B*K
#define DF   128
#define WWIN 25
#define NBLK 75     // 4800/64 tiles per dimension
#define RG   300    // 4800/16 row groups
#define LPAD 264    // padded LDS row stride in halves (528B = 132 dwords)

#define USE_ASYNC_LDS 1

// ---------------- K0: f32 -> f16 with 1/256 scaling -----------------
__global__ __launch_bounds__(256) void convert_kernel(const float* __restrict__ md0,
                                                      const float* __restrict__ md1,
                                                      _Float16* __restrict__ qh,
                                                      _Float16* __restrict__ rh, int n) {
  int i = blockIdx.x * 256 + threadIdx.x;
  if (i < n) {
    qh[i] = (_Float16)(md0[i] * (1.0f / 256.0f));
    rh[i] = (_Float16)(md1[i] * (1.0f / 256.0f));
  }
}

// ---------------- K1: sim GEMM (WMMA) -------------------------------
// Block: 64x64 tile, 4 waves (wave w -> rows 16w..16w+15).
// A (64x256) and B (64x256) panels async-copied to LDS once, K loop runs
// entirely from LDS. Epilogue: fused per-row / per-col partial softmax stats.
__global__ __launch_bounds__(128) void sim_gemm(const _Float16* __restrict__ qh,
                                                const _Float16* __restrict__ rh,
                                                float* __restrict__ out,
                                                float* __restrict__ rowPM,
                                                float* __restrict__ rowPS,
                                                float* __restrict__ colPM,
                                                float* __restrict__ colPS) {
  const int lane  = threadIdx.x & 31;
  const int wave  = threadIdx.x >> 5;
  const int b     = blockIdx.z;
  const int nblk  = blockIdx.x;
  const int mblk  = blockIdx.y;
  const int tileM = mblk * 64;
  const int tileN = nblk * 64;
  const int waveM = tileM + wave * 16;
  const int l15   = lane & 15;
  const int half  = lane >> 4;
  const int khalf = half * 8;       // A/B: K sub-block; C: M offset

  __shared__ _Float16 sA[64 * LPAD];
  __shared__ _Float16 sB[64 * LPAD];

  // ---- cooperative panel staging: 64 rows x 512B each, 16B chunks ----
  {
    const _Float16* gA = qh + ((size_t)(b * LDIM + tileM)) * DDIM;
    const _Float16* gB = rh + ((size_t)(b * LDIM + tileN)) * DDIM;
#pragma unroll
    for (int i = 0; i < 16; ++i) {
      const int c   = threadIdx.x + 128 * i;   // 0..2047 chunks per panel
      const int row = c >> 5;
      const int off = (c & 31) * 8;            // halves within row
#if USE_ASYNC_LDS
      unsigned ldsA = (unsigned)(size_t)&sA[row * LPAD + off];
      unsigned ldsB = (unsigned)(size_t)&sB[row * LPAD + off];
      unsigned long long gaA = (unsigned long long)(size_t)(gA + (size_t)row * DDIM + off);
      unsigned long long gaB = (unsigned long long)(size_t)(gB + (size_t)row * DDIM + off);
      asm volatile("global_load_async_to_lds_b128 %0, %1, off" :: "v"(ldsA), "v"(gaA) : "memory");
      asm volatile("global_load_async_to_lds_b128 %0, %1, off" :: "v"(ldsB), "v"(gaB) : "memory");
#else
      *(v8h*)&sA[row * LPAD + off] = *(const v8h*)(gA + (size_t)row * DDIM + off);
      *(v8h*)&sB[row * LPAD + off] = *(const v8h*)(gB + (size_t)row * DDIM + off);
#endif
    }
#if USE_ASYNC_LDS
    asm volatile("s_wait_asynccnt 0x0" ::: "memory");
#endif
    __syncthreads();
  }

  union Frag { v16h v; v8h h[2]; };
  v8f acc[4] = {v8f{}, v8f{}, v8f{}, v8f{}};

  const _Float16* sArow = sA + (wave * 16 + l15) * LPAD + khalf;
  const _Float16* sB0   = sB + l15 * LPAD + khalf;

#pragma unroll
  for (int kk = 0; kk < 8; ++kk) {
    const int k0 = kk * 32;
    Frag a;
    a.h[0] = *(const v8h*)(sArow + k0);
    a.h[1] = *(const v8h*)(sArow + k0 + 16);
#pragma unroll
    for (int j = 0; j < 4; ++j) {
      Frag bf;
      const _Float16* bb = sB0 + j * 16 * LPAD + k0;
      bf.h[0] = *(const v8h*)bb;
      bf.h[1] = *(const v8h*)(bb + 16);
      acc[j] = __builtin_amdgcn_wmma_f32_16x16x32_f16(
          false, a.v, false, bf.v, (short)0, acc[j], false, false);
    }
  }

  // scale by 1/temperature and store sim
  const size_t obase = (size_t)b * LDIM * LDIM;
  const int mrow = waveM + khalf;    // C layout: M = v + 8*(lane>=16)
#pragma unroll
  for (int j = 0; j < 4; ++j) {
#pragma unroll
    for (int v = 0; v < 8; ++v) acc[j][v] *= 10.0f;
    const int n = tileN + j * 16 + l15;
#pragma unroll
    for (int v = 0; v < 8; ++v)
      out[obase + (size_t)(mrow + v) * LDIM + n] = acc[j][v];
  }

  // ---- row partial stats: per row (16/wave), over this wave's 64 cols ----
#pragma unroll
  for (int v = 0; v < 8; ++v) {
    float pm = fmaxf(fmaxf(acc[0][v], acc[1][v]), fmaxf(acc[2][v], acc[3][v]));
    pm = fmaxf(pm, __shfl_xor(pm, 1));
    pm = fmaxf(pm, __shfl_xor(pm, 2));
    pm = fmaxf(pm, __shfl_xor(pm, 4));
    pm = fmaxf(pm, __shfl_xor(pm, 8));
    float ps = __expf(acc[0][v] - pm) + __expf(acc[1][v] - pm) +
               __expf(acc[2][v] - pm) + __expf(acc[3][v] - pm);
    ps += __shfl_xor(ps, 1);
    ps += __shfl_xor(ps, 2);
    ps += __shfl_xor(ps, 4);
    ps += __shfl_xor(ps, 8);
    if (l15 == v) {
      const int row = waveM + v + 8 * half;
      rowPM[((size_t)(b * LDIM + row)) * NBLK + nblk] = pm;
      rowPS[((size_t)(b * LDIM + row)) * NBLK + nblk] = ps;
    }
  }

  // ---- col partial stats: per col (64/block), over this block's 64 rows --
  __shared__ float sPM[4][64], sPS[4][64];
#pragma unroll
  for (int j = 0; j < 4; ++j) {
    float cmx = acc[j][0];
#pragma unroll
    for (int v = 1; v < 8; ++v) cmx = fmaxf(cmx, acc[j][v]);
    cmx = fmaxf(cmx, __shfl_xor(cmx, 16));       // combine the two 8-row halves
    float cs = 0.0f;
#pragma unroll
    for (int v = 0; v < 8; ++v) cs += __expf(acc[j][v] - cmx);
    cs += __shfl_xor(cs, 16);
    if (half == 0) { sPM[wave][j * 16 + l15] = cmx; sPS[wave][j * 16 + l15] = cs; }
  }
  __syncthreads();
  if (threadIdx.x < 64) {
    const int c = threadIdx.x;
    float m = fmaxf(fmaxf(sPM[0][c], sPM[1][c]), fmaxf(sPM[2][c], sPM[3][c]));
    float s = sPS[0][c] * __expf(sPM[0][c] - m) + sPS[1][c] * __expf(sPM[1][c] - m) +
              sPS[2][c] * __expf(sPM[2][c] - m) + sPS[3][c] * __expf(sPM[3][c] - m);
    colPM[((size_t)(b * LDIM + tileN + c)) * NBLK + mblk] = m;
    colPS[((size_t)(b * LDIM + tileN + c)) * NBLK + mblk] = s;
  }
}

// ---------------- K2: merge 75 partial (max,sum) pairs --------------
__global__ __launch_bounds__(256) void stat_combine(const float* __restrict__ pm,
                                                    const float* __restrict__ ps,
                                                    float* __restrict__ omax,
                                                    float* __restrict__ osum) {
  const int i = blockIdx.x * 256 + threadIdx.x;
  if (i >= 2 * LDIM) return;
  const float* pmr = pm + (size_t)i * NBLK;
  const float* psr = ps + (size_t)i * NBLK;
  float m = -INFINITY, s = 0.0f;
  for (int p = 0; p < NBLK; ++p) {
    float m2 = pmr[p], s2 = psr[p];
    float nm = fmaxf(m, m2);
    s = s * __expf(m - nm) + s2 * __expf(m2 - nm);
    m = nm;
  }
  omax[i] = m;
  osum[i] = s;
}

// ---------------- K3: cm in place + row max/argmax + col-max partials
__global__ __launch_bounds__(256) void cm_pass(float* __restrict__ cm,
                                               const float* __restrict__ rmaxv,
                                               const float* __restrict__ rsumv,
                                               const float* __restrict__ cmaxv,
                                               const float* __restrict__ csumv,
                                               float* __restrict__ rmax2,
                                               int* __restrict__ rarg,
                                               float* __restrict__ colpart) {
  const int rg  = blockIdx.x;      // 0..299
  const int b   = blockIdx.y;
  const int tid = threadIdx.x;
  float rm[16], ris[16], rbest[16];
  int rbidx[16];
#pragma unroll
  for (int r = 0; r < 16; ++r) {
    const int bl = b * LDIM + rg * 16 + r;
    rm[r]   = rmaxv[bl];
    ris[r]  = 1.0f / rsumv[bl];
    rbest[r] = -INFINITY;
    rbidx[r] = 0;
  }
  for (int base = 0; base < LDIM; base += 256) {
    const int col = base + tid;
    if (col < LDIM) {
      if (col + 256 < LDIM)
        __builtin_prefetch(&cm[((size_t)(b * LDIM + rg * 16)) * LDIM + col + 256], 0, 1);
      const float cmx = cmaxv[b * LDIM + col];
      const float ics = 1.0f / csumv[b * LDIM + col];
      float cpm = -INFINITY;
#pragma unroll
      for (int r = 0; r < 16; ++r) {
        const size_t idx = ((size_t)(b * LDIM + rg * 16 + r)) * LDIM + col;
        const float x = cm[idx];
        const float v = __expf(2.0f * x - rm[r] - cmx) * ris[r] * ics;
        cm[idx] = v;
        if (v > rbest[r]) { rbest[r] = v; rbidx[r] = col; }
        cpm = fmaxf(cpm, v);
      }
      colpart[((size_t)(b * LDIM + col)) * RG + rg] = cpm;
    }
  }
  __shared__ float rv[256];
  __shared__ int ri[256];
  for (int r = 0; r < 16; ++r) {
    rv[tid] = rbest[r]; ri[tid] = rbidx[r];
    __syncthreads();
    for (int off = 128; off; off >>= 1) {
      if (tid < off) {
        float v2 = rv[tid + off]; int i2 = ri[tid + off];
        if (v2 > rv[tid] || (v2 == rv[tid] && i2 < ri[tid])) { rv[tid] = v2; ri[tid] = i2; }
      }
      __syncthreads();
    }
    if (tid == 0) { const int bl = b * LDIM + rg * 16 + r; rmax2[bl] = rv[0]; rarg[bl] = ri[0]; }
    __syncthreads();
  }
}

// ---------------- K4: merge 300 per-column cm-max partials ----------
__global__ __launch_bounds__(256) void colmax_combine(const float* __restrict__ colpart,
                                                      float* __restrict__ cmax2) {
  const int i = blockIdx.x * 256 + threadIdx.x;
  if (i >= 2 * LDIM) return;
  const float* p = colpart + (size_t)i * RG;
  float m = -INFINITY;
  for (int r = 0; r < RG; ++r) m = fmaxf(m, p[r]);
  cmax2[i] = m;
}

// ---------------- K5: mutual-NN mask --------------------------------
__global__ __launch_bounds__(256) void mutual_mask(const float* __restrict__ rmax2,
                                                   const int* __restrict__ rarg,
                                                   const float* __restrict__ cmax2,
                                                   float* __restrict__ maskv,
                                                   int* __restrict__ allj) {
  const int bl = blockIdx.x * 256 + threadIdx.x;
  if (bl >= 2 * LDIM) return;
  const int b = bl / LDIM;
  const int j = rarg[bl];
  const float v = rmax2[bl];
  const bool ok = (v == cmax2[b * LDIM + j]);
  maskv[bl] = ok ? v : 0.0f;
  allj[bl]  = ok ? j : 0;
}

// ---------------- K6: top-512 per batch (iterative argmax) ----------
__global__ __launch_bounds__(256) void topk_kernel(const float* __restrict__ maskv,
                                                   const int* __restrict__ allj,
                                                   int* __restrict__ iids,
                                                   int* __restrict__ jids,
                                                   float* __restrict__ basexy) {
  const int b = blockIdx.x;
  __shared__ float vals[LDIM];
  __shared__ float rv[256];
  __shared__ int ri[256];
  for (int i = threadIdx.x; i < LDIM; i += 256) vals[i] = maskv[b * LDIM + i];
  __syncthreads();
  for (int k = 0; k < KTOP; ++k) {
    float bv = -INFINITY; int bi = 0x7fffffff;
    for (int i = threadIdx.x; i < LDIM; i += 256) {
      float v = vals[i];
      if (v > bv) { bv = v; bi = i; }
    }
    rv[threadIdx.x] = bv; ri[threadIdx.x] = bi;
    __syncthreads();
    for (int off = 128; off; off >>= 1) {
      if ((int)threadIdx.x < off) {
        float v2 = rv[threadIdx.x + off]; int i2 = ri[threadIdx.x + off];
        if (v2 > rv[threadIdx.x] || (v2 == rv[threadIdx.x] && i2 < ri[threadIdx.x])) {
          rv[threadIdx.x] = v2; ri[threadIdx.x] = i2;
        }
      }
      __syncthreads();
    }
    if (threadIdx.x == 0) {
      const int ii = ri[0];
      iids[b * KTOP + k] = ii;
      const int j = allj[b * LDIM + ii];
      jids[b * KTOP + k] = j;
      const int t = b * KTOP + k;
      basexy[t * 2 + 0] = (float)((j % 80) * 8);
      basexy[t * 2 + 1] = (float)((j / 80) * 8);
      vals[ii] = -INFINITY;
    }
    __syncthreads();
  }
}

// ---------------- K7: center = mdesc[sel] @ proj_w + proj_b ---------
__global__ __launch_bounds__(128) void center_proj(const float* __restrict__ md0,
                                                   const float* __restrict__ md1,
                                                   const int* __restrict__ iids,
                                                   const int* __restrict__ jids,
                                                   const float* __restrict__ pw,
                                                   const float* __restrict__ pb,
                                                   float* __restrict__ center) {
  const int t = blockIdx.x;   // 0 .. 2*NPTS-1
  const float* src; int b, row;
  if (t < NPTS) { b = t / KTOP; row = iids[t]; src = md0; }
  else { const int u = t - NPTS; b = u / KTOP; row = jids[u]; src = md1; }
  __shared__ float cv[DDIM];
  const float* r = src + ((size_t)b * LDIM + row) * DDIM;
  for (int i = threadIdx.x; i < DDIM; i += 128) cv[i] = r[i];
  __syncthreads();
  const int o = threadIdx.x;
  float s = pb[o];
  for (int d = 0; d < DDIM; ++d) s += cv[d] * pw[d * DF + o];
  center[t * DF + o] = s;
}

// ---------------- K8: unfold-gather + merge MLP + positional enc ----
__global__ __launch_bounds__(128) void merge_kernel(const float* __restrict__ f0,
                                                    const float* __restrict__ f1,
                                                    const int* __restrict__ iids,
                                                    const int* __restrict__ jids,
                                                    const float* __restrict__ center,
                                                    const float* __restrict__ mw,
                                                    const float* __restrict__ mb,
                                                    float* __restrict__ localb) {
  const int w = blockIdx.x;   // 0..24
  const int t = blockIdx.y;   // 0..2*NPTS-1
  int b, idx; const float* feat;
  if (t < NPTS) { b = t / KTOP; idx = iids[t]; feat = f0; }
  else { const int u = t - NPTS; b = u / KTOP; idx = jids[u]; feat = f1; }
  const int ho = idx / 80, wo = idx % 80;
  const int dy = w / 5, dx = w % 5;
  const int y = 4 * ho + dy - 2, x = 4 * wo + dx - 2;
  __shared__ float lv[DF], cvv[DF];
  const int c = threadIdx.x;
  float v = 0.0f;
  if (y >= 0 && y < 240 && x >= 0 && x < 320)
    v = feat[(((size_t)b * DF + c) * 240 + y) * 320 + x];
  lv[c]  = v;
  cvv[c] = center[t * DF + c];
  __syncthreads();
  const int o = c;
  float s = mb[o];
  for (int d = 0; d < DF; ++d) {
    s += lv[d]  * mw[d * DF + o];
    s += cvv[d] * mw[(DF + d) * DF + o];
  }
  const float e   = __expf(-logf(10000.0f) * (float)(o & ~1) / (float)DF);
  const float ang = (float)w * e;
  s += (o & 1) ? cosf(ang) : sinf(ang);
  localb[((size_t)t * WWIN + w) * DF + o] = s;
}

// ---------------- K9: reg1 MLP --------------------------------------
__global__ __launch_bounds__(128) void reg1_kernel(const float* __restrict__ localb,
                                                   const float* __restrict__ w1,
                                                   const float* __restrict__ b1,
                                                   float* __restrict__ featb) {
  const int w = blockIdx.x;   // 0..24
  const int t = blockIdx.y;   // 0..NPTS-1
  __shared__ float cen[DF], fd[DF];
  const int c = threadIdx.x;
  cen[c] = localb[((size_t)t * WWIN + 12) * DF + c];
  fd[c]  = localb[(((size_t)(NPTS + t)) * WWIN + w) * DF + c];
  __syncthreads();
  const int o = c;
  float s = b1[o];
  for (int d = 0; d < DF; ++d) {
    s += cen[d] * w1[d * DF + o];
    s += fd[d]  * w1[(DF + d) * DF + o];
  }
  featb[(size_t)t * (WWIN * DF) + w * DF + o] = s;
}

// ---------------- K10: reg2 MLP (3200 -> 128) -----------------------
__global__ __launch_bounds__(128) void reg2_kernel(const float* __restrict__ featb,
                                                   const float* __restrict__ w2,
                                                   const float* __restrict__ b2,
                                                   float* __restrict__ out2) {
  const int t = blockIdx.x;
  const int o = threadIdx.x;
  __shared__ float chunk[128];
  float s = b2[o];
  const float* fr = featb + (size_t)t * (WWIN * DF);
  for (int base = 0; base < WWIN * DF; base += 128) {
    __syncthreads();
    chunk[threadIdx.x] = fr[base + threadIdx.x];
    __syncthreads();
    for (int d = 0; d < 128; ++d) s += chunk[d] * w2[(size_t)(base + d) * DF + o];
  }
  out2[t * DF + o] = s;
}

// ---------------- K11: coords + mkpts1 ------------------------------
__global__ __launch_bounds__(256) void final_kernel(const float* __restrict__ out2,
                                                    const float* __restrict__ rw,
                                                    const float* __restrict__ rb,
                                                    const float* __restrict__ basexy,
                                                    float* __restrict__ mk) {
  const int i = blockIdx.x * 256 + threadIdx.x;
  if (i >= NPTS * 2) return;
  const int t = i >> 1, c = i & 1;
  float s = rb[c];
  for (int o = 0; o < DF; ++o) s += out2[t * DF + o] * rw[o * 2 + c];
  mk[i] = basexy[i] + s;
}

extern "C" void kernel_launch(void* const* d_in, const int* in_sizes, int n_in,
                              void* d_out, int out_size, void* d_ws, size_t ws_size,
                              hipStream_t stream) {
  const float* mdesc0 = (const float*)d_in[3];
  const float* mdesc1 = (const float*)d_in[4];
  const float* feat0  = (const float*)d_in[5];
  const float* feat1  = (const float*)d_in[6];
  const float* proj_w = (const float*)d_in[7];
  const float* proj_b = (const float*)d_in[8];
  const float* merge_w = (const float*)d_in[9];
  const float* merge_b = (const float*)d_in[10];
  const float* reg1_w = (const float*)d_in[11];
  const float* reg1_b = (const float*)d_in[12];
  const float* reg2_w = (const float*)d_in[13];
  const float* reg2_b = (const float*)d_in[14];
  const float* reg_w  = (const float*)d_in[15];
  const float* reg_b  = (const float*)d_in[16];

  float* cm = (float*)d_out;                 // (2,4800,4800)
  float* mk = cm + (size_t)2 * LDIM * LDIM;  // (1024,2)

  char* ws = (char*)d_ws;
  size_t off = 0;
  auto alloc = [&](size_t bytes) {
    char* p = ws + off;
    off += (bytes + 255) & ~(size_t)255;
    return p;
  };
  const int nmd = 2 * LDIM * DDIM;
  _Float16* qh  = (_Float16*)alloc((size_t)nmd * 2);
  _Float16* rh  = (_Float16*)alloc((size_t)nmd * 2);
  float* rowPM  = (float*)alloc((size_t)2 * LDIM * NBLK * 4);
  float* rowPS  = (float*)alloc((size_t)2 * LDIM * NBLK * 4);
  float* colPM  = (float*)alloc((size_t)2 * LDIM * NBLK * 4);
  float* colPS  = (float*)alloc((size_t)2 * LDIM * NBLK * 4);
  float* colpart = (float*)alloc((size_t)2 * LDIM * RG * 4);
  float* rmaxv  = (float*)alloc(2 * LDIM * 4);
  float* rsumv  = (float*)alloc(2 * LDIM * 4);
  float* cmaxv  = (float*)alloc(2 * LDIM * 4);
  float* csumv  = (float*)alloc(2 * LDIM * 4);
  float* rmax2  = (float*)alloc(2 * LDIM * 4);
  int*   rarg   = (int*)alloc(2 * LDIM * 4);
  float* cmax2  = (float*)alloc(2 * LDIM * 4);
  float* maskv  = (float*)alloc(2 * LDIM * 4);
  int*   allj   = (int*)alloc(2 * LDIM * 4);
  int*   iids   = (int*)alloc(NPTS * 4);
  int*   jids   = (int*)alloc(NPTS * 4);
  float* basexy = (float*)alloc(NPTS * 2 * 4);
  float* center = (float*)alloc((size_t)2 * NPTS * DF * 4);
  float* localb = (float*)alloc((size_t)2 * NPTS * WWIN * DF * 4);
  float* featb  = (float*)alloc((size_t)NPTS * WWIN * DF * 4);
  float* out2   = (float*)alloc((size_t)NPTS * DF * 4);
  (void)ws_size; (void)in_sizes; (void)n_in; (void)out_size;

  convert_kernel<<<(nmd + 255) / 256, 256, 0, stream>>>(mdesc0, mdesc1, qh, rh, nmd);

  sim_gemm<<<dim3(NBLK, NBLK, 2), 128, 0, stream>>>(qh, rh, cm, rowPM, rowPS, colPM, colPS);

  stat_combine<<<(2 * LDIM + 255) / 256, 256, 0, stream>>>(rowPM, rowPS, rmaxv, rsumv);
  stat_combine<<<(2 * LDIM + 255) / 256, 256, 0, stream>>>(colPM, colPS, cmaxv, csumv);

  cm_pass<<<dim3(RG, 2), 256, 0, stream>>>(cm, rmaxv, rsumv, cmaxv, csumv, rmax2, rarg, colpart);

  colmax_combine<<<(2 * LDIM + 255) / 256, 256, 0, stream>>>(colpart, cmax2);

  mutual_mask<<<(2 * LDIM + 255) / 256, 256, 0, stream>>>(rmax2, rarg, cmax2, maskv, allj);

  topk_kernel<<<2, 256, 0, stream>>>(maskv, allj, iids, jids, basexy);

  center_proj<<<2 * NPTS, 128, 0, stream>>>(mdesc0, mdesc1, iids, jids, proj_w, proj_b, center);

  merge_kernel<<<dim3(WWIN, 2 * NPTS), 128, 0, stream>>>(feat0, feat1, iids, jids, center,
                                                         merge_w, merge_b, localb);

  reg1_kernel<<<dim3(WWIN, NPTS), 128, 0, stream>>>(localb, reg1_w, reg1_b, featb);

  reg2_kernel<<<NPTS, 128, 0, stream>>>(featb, reg2_w, reg2_b, out2);

  final_kernel<<<(NPTS * 2 + 255) / 256, 256, 0, stream>>>(out2, reg_w, reg_b, basexy, mk);
}
